// Attention_48808008352252
// MI455X (gfx1250) — matrix operand order repaired and checked by reading, not compiled, not run on hardware
//
#include <hip/hip_runtime.h>

#define HEADS 16
#define DHEAD 64
#define NSEQ 2048
#define BATCH 2
#define DMODEL 1024
#define INNER 1024
#define ATT_SCALE 0.125f  // 64^-0.5

// LDS layout for the attention kernel (bytes)
#define KSTRIDE 160       // 64 bf16 (128B) + 32B pad -> 32B aligned rows, few bank conflicts
#define VSTRIDE 96        // 32 bf16 (64B) + 32B pad
#define KBUF0   0
#define KBUF1   5120      // 32*160
#define VBUF0   10240
#define VBUF1   16384     // +64*96
#define PLSOFF  22528
#define SMEMSZ  26624     // + 4 waves * 1KB P tiles

typedef __attribute__((ext_vector_type(16))) __bf16 v16bf;
typedef __attribute__((ext_vector_type(8)))  __bf16 v8bf;
typedef __attribute__((ext_vector_type(8)))  float  v8f;

union V16U { v16bf v; v8bf h[2]; };

static __device__ __forceinline__ unsigned short f2bf(float f) {
    unsigned int u = __float_as_uint(f);
    u += 0x7FFFu + ((u >> 16) & 1u);   // round-to-nearest-even
    return (unsigned short)(u >> 16);
}

// A-fragment: two adjacent 16-byte chunks per lane (k = 16*half + s)
static __device__ __forceinline__ v16bf ldfrag(const unsigned short* p0,
                                               const unsigned short* p1) {
    V16U u;
    u.h[0] = *(const v8bf*)p0;
    u.h[1] = *(const v8bf*)p1;
    return u.v;
}

static __device__ __forceinline__ v8f wmma_bf16(v16bf a, v16bf b, v8f c) {
    return __builtin_amdgcn_wmma_f32_16x16x32_bf16(
        /*neg_a=*/false, a, /*neg_b=*/false, b,
        /*c_mod=*/(short)0, c, /*reuse_a=*/false, /*reuse_b=*/false);
}

static __device__ __forceinline__ float redmax16(float v) {
    v = fmaxf(v, __shfl_xor(v, 1, 32));
    v = fmaxf(v, __shfl_xor(v, 2, 32));
    v = fmaxf(v, __shfl_xor(v, 4, 32));
    v = fmaxf(v, __shfl_xor(v, 8, 32));
    return v;
}
static __device__ __forceinline__ float redsum16(float v) {
    v += __shfl_xor(v, 1, 32);
    v += __shfl_xor(v, 2, 32);
    v += __shfl_xor(v, 4, 32);
    v += __shfl_xor(v, 8, 32);
    return v;
}

// CDNA5 async global->LDS copy, 16B per lane, tracked by ASYNCcnt (ISA 08)
static __device__ __forceinline__ void async_cp16(unsigned ldsoff, const void* g) {
    asm volatile("global_load_async_to_lds_b128 %0, %1, off"
                 :: "v"(ldsoff), "v"((unsigned long long)(uintptr_t)g)
                 : "memory");
}

// 128 lanes cooperatively stage K tile (32 keys x 64 dims) and V tile (64 dims x 32 keys)
static __device__ __forceinline__ void stage_tiles(unsigned kdst, unsigned vdst,
        const unsigned short* Kh, const unsigned short* Vh, int j0, int tid) {
    #pragma unroll
    for (int i = 0; i < 2; i++) {
        const int c = tid + i * 128;          // 0..255: 16B chunk id
        const int krow = c >> 3, kp = c & 7;  // K: 32 rows x 8 chunks
        async_cp16(kdst + krow * KSTRIDE + kp * 16,
                   (const char*)Kh + (size_t)(j0 + krow) * 128 + kp * 16);
        const int vrow = c >> 2, vp = c & 3;  // V: 64 rows x 4 chunks
        async_cp16(vdst + vrow * VSTRIDE + vp * 16,
                   (const char*)Vh + (size_t)vrow * (NSEQ * 2) + (size_t)j0 * 2 + vp * 16);
    }
}

// ---------------- prep: cast x to bf16 ----------------
__global__ void k_prep_x(const float* __restrict__ x, unsigned short* __restrict__ xb) {
    int i = blockIdx.x * 256 + threadIdx.x;
    xb[i] = f2bf(x[i]);
}

// ------- prep: transpose + cast the 4 weight matrices (Wt[n][k] = W[k][n]) -------
__global__ void k_prep_w(const float* __restrict__ wq, const float* __restrict__ wk,
                         const float* __restrict__ wv, const float* __restrict__ wo,
                         unsigned short* __restrict__ tq, unsigned short* __restrict__ tk,
                         unsigned short* __restrict__ tv, unsigned short* __restrict__ to_) {
    int i = blockIdx.x * 256 + threadIdx.x;     // 0 .. 1M-1
    int n = i >> 10, k = i & 1023;
    const float* s;
    unsigned short* d;
    switch (blockIdx.y) {
        case 0:  s = wq; d = tq;  break;
        case 1:  s = wk; d = tk;  break;
        case 2:  s = wv; d = tv;  break;
        default: s = wo; d = to_; break;
    }
    d[i] = f2bf(s[(size_t)k * 1024 + n]);
}

// ---------------- Q/K/V projection GEMM (one wave = 16x64 output tile) ----------------
// mode (blockIdx.z): 0 -> Q (scaled, [b,h,n,d]), 1 -> K ([b,h,n,d]), 2 -> V transposed ([b,h,d,n])
__global__ __launch_bounds__(32) void k_proj(const unsigned short* __restrict__ xb,
        const unsigned short* __restrict__ wqt, const unsigned short* __restrict__ wkt,
        const unsigned short* __restrict__ wvt,
        unsigned short* __restrict__ Qb, unsigned short* __restrict__ Kb,
        unsigned short* __restrict__ Vt) {
    const int lane = threadIdx.x;
    const int half = lane >> 4;
    const int lr   = lane & 15;
    const int mbase = blockIdx.x * 16;
    const int nbase = blockIdx.y * 64;
    const int mode  = blockIdx.z;
    const unsigned short* wt = (mode == 0) ? wqt : (mode == 1) ? wkt : wvt;

    v8f c[4] = {};
    const unsigned short* arow = xb + (size_t)(mbase + lr) * INNER;
    for (int k = 0; k < DMODEL; k += 32) {
        v16bf a = ldfrag(arow + k + half * 16, arow + k + half * 16 + 8);
        #pragma unroll
        for (int t = 0; t < 4; t++) {
            const unsigned short* bp =
                wt + (size_t)(nbase + t * 16 + lr) * DMODEL + k + half * 16;
            c[t] = wmma_bf16(a, *(const v16bf*)bp, c[t]);
        }
    }
    const float sc = (mode == 0) ? ATT_SCALE : 1.0f;
    #pragma unroll
    for (int t = 0; t < 4; t++) {
        const int col = nbase + t * 16 + lr;
        const int h = col >> 6, dh = col & 63;
        #pragma unroll
        for (int v = 0; v < 8; v++) {
            const int row = mbase + v + half * 8;
            const int b = row >> 11, n = row & 2047;
            unsigned short bv = f2bf(c[t][v] * sc);
            if (mode == 2)
                Vt[((size_t)(b * HEADS + h) * DHEAD + dh) * NSEQ + n] = bv;
            else if (mode == 1)
                Kb[((size_t)(b * HEADS + h) * NSEQ + n) * DHEAD + dh] = bv;
            else
                Qb[((size_t)(b * HEADS + h) * NSEQ + n) * DHEAD + dh] = bv;
        }
    }
}

// -------- flash attention: 4 waves / block, 64 query rows, async double-buffered K/V --------
__global__ __launch_bounds__(128) void k_attn(const unsigned short* __restrict__ Qb,
        const unsigned short* __restrict__ Kb, const unsigned short* __restrict__ Vt,
        unsigned short* __restrict__ Ob) {
    __shared__ __align__(128) char smem[SMEMSZ];
    const int tid  = threadIdx.x;
    const int wid  = tid >> 5;
    const int lane = tid & 31;
    const int half = lane >> 4;
    const int lr   = lane & 15;
    const int q0 = blockIdx.x * 64 + wid * 16;
    const int h  = blockIdx.y;
    const int b  = blockIdx.z;
    const size_t headOff = (size_t)(b * HEADS + h) * NSEQ * DHEAD;
    const unsigned short* Qh = Qb + headOff;   // [2048][64]
    const unsigned short* Kh = Kb + headOff;   // [2048][64]
    const unsigned short* Vh = Vt + headOff;   // [64][2048]

    const unsigned ubase = (unsigned)(uintptr_t)(void*)smem;  // LDS byte offset of smem
    unsigned short* pls = (unsigned short*)(smem + PLSOFF + wid * 1024);

    const unsigned short* qrow = Qh + (size_t)(q0 + lr) * DHEAD;
    v16bf aq0 = ldfrag(qrow +      half * 16, qrow +  8 + half * 16);  // kdim 0..31
    v16bf aq1 = ldfrag(qrow + 32 + half * 16, qrow + 40 + half * 16);  // kdim 32..63

    v8f o[4] = {};
    float m[8], l[8];
    #pragma unroll
    for (int v = 0; v < 8; v++) { m[v] = -1e30f; l[v] = 0.0f; }

    // prefetch first key block into buffer 0
    stage_tiles(ubase + KBUF0, ubase + VBUF0, Kh, Vh, 0, tid);

    for (int j0 = 0; j0 < NSEQ; j0 += 32) {
        const int cur = (j0 >> 5) & 1;
        asm volatile("s_wait_asynccnt 0" ::: "memory");   // own prefetch landed
        __syncthreads();                                   // all slices landed; prev readers done
        if (j0 + 32 < NSEQ)                                // prefetch next block into other buffer
            stage_tiles(ubase + (cur ? KBUF0 : KBUF1),
                        ubase + (cur ? VBUF0 : VBUF1), Kh, Vh, j0 + 32, tid);

        const char* kcur = smem + (cur ? KBUF1 : KBUF0);
        const char* vcur = smem + (cur ? VBUF1 : VBUF0);

        // scores: two 16-query x 16-key tiles, K B-fragments from LDS
        v8f s0 = {}, s1 = {};
        {
            const char* kr0 = kcur + (size_t)lr * KSTRIDE + half * 32;
            const char* kr1 = kcur + (size_t)(16 + lr) * KSTRIDE + half * 32;
            s0 = wmma_bf16(aq0, *(const v16bf*)kr0,        s0);
            s0 = wmma_bf16(aq1, *(const v16bf*)(kr0 + 64), s0);
            s1 = wmma_bf16(aq0, *(const v16bf*)kr1,        s1);
            s1 = wmma_bf16(aq1, *(const v16bf*)(kr1 + 64), s1);
        }

        // streaming softmax (row stats shared across the 16 lanes of each half)
        float alpha[8];
        #pragma unroll
        for (int v = 0; v < 8; v++) {
            float mx = redmax16(fmaxf(s0[v], s1[v]));
            float mn = fmaxf(m[v], mx);
            alpha[v] = __expf(m[v] - mn);
            float p0 = __expf(s0[v] - mn);
            float p1 = __expf(s1[v] - mn);
            s0[v] = p0; s1[v] = p1;
            l[v] = l[v] * alpha[v] + redsum16(p0 + p1);
            m[v] = mn;
        }
        #pragma unroll
        for (int t = 0; t < 4; t++) {
            #pragma unroll
            for (int v = 0; v < 8; v++) { o[t][v] *= alpha[v]; }
        }

        // C/D-layout P -> row-major LDS tile (per-wave private slice)
        #pragma unroll
        for (int v = 0; v < 8; v++) {
            const int r = v + half * 8;
            pls[r * 32 + lr]      = f2bf(s0[v]);
            pls[r * 32 + 16 + lr] = f2bf(s1[v]);
        }
        asm volatile("s_wait_dscnt 0" ::: "memory");   // same-wave LDS is in-order; pin compiler

        // re-read P as an A-fragment (16x32)
        const unsigned short* pr = pls + lr * 32;
        v16bf ap = ldfrag(pr + half * 16, pr + half * 16 + 8);

        // accumulate O += P * V, V B-fragments from LDS
        #pragma unroll
        for (int t = 0; t < 4; t++) {
            const char* vr = vcur + (size_t)(t * 16 + lr) * VSTRIDE + half * 32;
            o[t] = wmma_bf16(ap, *(const v16bf*)vr, o[t]);
        }
        asm volatile("s_wait_dscnt 0" ::: "memory");
    }

    // normalize and store O as bf16 [4096][1024]
    #pragma unroll
    for (int t = 0; t < 4; t++) {
        const int col = h * DHEAD + t * 16 + lr;
        #pragma unroll
        for (int v = 0; v < 8; v++) {
            const int r = v + half * 8;
            const size_t row = (size_t)b * NSEQ + q0 + r;
            Ob[row * INNER + col] = f2bf(o[t][v] / l[v]);
        }
    }
}

// ---------------- output GEMM: out = O * Wo + bo (f32 result) ----------------
__global__ __launch_bounds__(32) void k_outgemm(const unsigned short* __restrict__ Ob,
        const unsigned short* __restrict__ Wot, const float* __restrict__ bo,
        float* __restrict__ out) {
    const int lane = threadIdx.x;
    const int half = lane >> 4;
    const int lr   = lane & 15;
    const int mbase = blockIdx.x * 16;
    const int nbase = blockIdx.y * 64;

    v8f c[4] = {};
    const unsigned short* arow = Ob + (size_t)(mbase + lr) * INNER;
    for (int k = 0; k < INNER; k += 32) {
        v16bf a = ldfrag(arow + k + half * 16, arow + k + half * 16 + 8);
        #pragma unroll
        for (int t = 0; t < 4; t++) {
            const unsigned short* bp =
                Wot + (size_t)(nbase + t * 16 + lr) * INNER + k + half * 16;
            c[t] = wmma_bf16(a, *(const v16bf*)bp, c[t]);
        }
    }
    #pragma unroll
    for (int t = 0; t < 4; t++) {
        const int col = nbase + t * 16 + lr;
        const float bias = bo[col];
        #pragma unroll
        for (int v = 0; v < 8; v++) {
            const int row = mbase + v + half * 8;
            out[(size_t)row * DMODEL + col] = c[t][v] + bias;
        }
    }
}

extern "C" void kernel_launch(void* const* d_in, const int* in_sizes, int n_in,
                              void* d_out, int out_size, void* d_ws, size_t ws_size,
                              hipStream_t stream) {
    const float* x  = (const float*)d_in[0];
    const float* Wq = (const float*)d_in[1];
    const float* Wk = (const float*)d_in[2];
    const float* Wv = (const float*)d_in[3];
    const float* Wo = (const float*)d_in[4];
    const float* bo = (const float*)d_in[5];
    float* out = (float*)d_out;

    char* ws = (char*)d_ws;
    const size_t MB = 1ull << 20;
    unsigned short* xb  = (unsigned short*)(ws + 0 * MB);   // 8 MiB  x in bf16
    unsigned short* Wqt = (unsigned short*)(ws + 8 * MB);   // 2 MiB  Wq^T bf16
    unsigned short* Wkt = (unsigned short*)(ws + 10 * MB);  // 2 MiB
    unsigned short* Wvt = (unsigned short*)(ws + 12 * MB);  // 2 MiB
    unsigned short* Wot = (unsigned short*)(ws + 14 * MB);  // 2 MiB
    unsigned short* Qb  = (unsigned short*)(ws + 16 * MB);  // 8 MiB  [b,h,n,d]
    unsigned short* Kb  = (unsigned short*)(ws + 24 * MB);  // 8 MiB  [b,h,n,d]
    unsigned short* Vt  = (unsigned short*)(ws + 32 * MB);  // 8 MiB  [b,h,d,n]
    unsigned short* Ob  = (unsigned short*)(ws + 40 * MB);  // 8 MiB  [4096,1024]

    k_prep_x<<<dim3(16384), dim3(256), 0, stream>>>(x, xb);
    k_prep_w<<<dim3(4096, 4), dim3(256), 0, stream>>>(Wq, Wk, Wv, Wo, Wqt, Wkt, Wvt, Wot);
    k_proj<<<dim3(256, 16, 3), dim3(32), 0, stream>>>(xb, Wqt, Wkt, Wvt, Qb, Kb, Vt);
    k_attn<<<dim3(32, 16, 2), dim3(128), 0, stream>>>(Qb, Kb, Vt, Ob);
    k_outgemm<<<dim3(256, 16), dim3(32), 0, stream>>>(Ob, Wot, bo, out);
}
